// LSTM_31894427140161
// MI455X (gfx1250) — matrix operand order repaired and checked by reading, not compiled, not run on hardware
//
#include <hip/hip_runtime.h>

// ---------------------------------------------------------------------------
// LSTM (L=2, B=64, T=256, I=H=1024) for gfx1250 using v_wmma_f32_16x16x32_f16.
// Weights converted once to f16 (32MB -> L2 resident), sequence walked with
// one fused 4-gate WMMA kernel per (t, layer); all launches on `stream` so
// the harness graph-capture amortizes the 512 step launches.
// ---------------------------------------------------------------------------

#define LSTM_B 64
#define LSTM_T 256
#define LSTM_I 1024
#define LSTM_H 1024
#define LSTM_K 2048   // I + H
#define LSTM_L 2

typedef __attribute__((ext_vector_type(16))) _Float16 v16h;
typedef __attribute__((ext_vector_type(8)))  _Float16 v8h;
typedef __attribute__((ext_vector_type(8)))  float    v8f;

// ---------------- prep kernels ----------------

__global__ void f32_to_f16_kernel(const float* __restrict__ src,
                                  _Float16* __restrict__ dst, int n) {
    for (int i = blockIdx.x * blockDim.x + threadIdx.x; i < n;
         i += gridDim.x * blockDim.x)
        dst[i] = (_Float16)src[i];
}

__global__ void zero_u32_kernel(unsigned* __restrict__ p, int n) {
    for (int i = blockIdx.x * blockDim.x + threadIdx.x; i < n;
         i += gridDim.x * blockDim.x)
        p[i] = 0u;
}

__global__ void copy_states_kernel(const float* __restrict__ hs,
                                   const float* __restrict__ cs,
                                   float* __restrict__ out, int n) {
    for (int i = blockIdx.x * blockDim.x + threadIdx.x; i < n;
         i += gridDim.x * blockDim.x) {
        out[i]     = hs[i];
        out[n + i] = cs[i];
    }
}

// ---------------- fused 4-gate LSTM step ----------------
//
// One workgroup = 4 waves = gates {f, i, chat, o} for one 16x16 (batch x H)
// output tile.  Each wave: 64 x v_wmma_f32_16x16x32_f16 over K=2048.
// A-operand (activations): rows = batch; first 1024 cols from aX (x_t or
// h0_t), last 1024 cols from hPrev.  B-operand: W[gate] rows (contiguous K).
//
// 16-bit A lane layout: lane m=lane&15, kh=lane>>4; per wmma the
// lane holds 16 contiguous K at kh*16 (same order as B).
// 16-bit B lane layout: lane n=lane&15 holds 16 contiguous K at kh*16.

__global__ __launch_bounds__(128)
void lstm_step_kernel(const _Float16* __restrict__ aX, size_t strideA,
                      const _Float16* __restrict__ hPrev,
                      const _Float16* __restrict__ Wl,   // +l*H*K, gate stride L*H*K
                      const float* __restrict__ bF, const float* __restrict__ bI,
                      const float* __restrict__ bC, const float* __restrict__ bO,
                      float* __restrict__ cState,        // [B,H] in/out
                      float* __restrict__ hOutF32,       // [B,H]
                      _Float16* __restrict__ hOutF16,    // [B,H]
                      float* __restrict__ seqOut,        // d_out (layer 1) or null
                      int t) {
    const int lane = threadIdx.x & 31;
    const int wave = threadIdx.x >> 5;      // gate id: 0=f 1=i 2=chat 3=o
    const int l15  = lane & 15;
    const int kh   = lane >> 4;
    const int nt   = blockIdx.x;            // 0..63  (H tiles)
    const int mt   = blockIdx.y;            // 0..3   (batch tiles)

    const int m = mt * 16 + l15;            // batch row this lane loads (A)
    const int n = nt * 16 + l15;            // out col  this lane loads (B)

    const _Float16* arow0 = aX    + (size_t)m * strideA;            // K 0..1023
    const _Float16* arow1 = hPrev + (size_t)m * LSTM_H;             // K 1024..2047
    const _Float16* brow  = Wl + (size_t)wave * (LSTM_L * (size_t)LSTM_H * LSTM_K)
                               + (size_t)n * LSTM_K;

    v8f acc = {};

#pragma unroll 8
    for (int kk = 0; kk < 32; ++kk) {       // K = 0..1023 (x / h0 part)
        const int ka = kk * 32 + kh * 16;
        v16h a  = *(const v16h*)(arow0 + ka);
        v16h b  = *(const v16h*)(brow + kk * 32 + kh * 16);
        acc = __builtin_amdgcn_wmma_f32_16x16x32_f16(false, a, false, b,
                                                     (short)0, acc, false, false);
    }
#pragma unroll 8
    for (int kk = 0; kk < 32; ++kk) {       // K = 1024..2047 (recurrent part)
        const int ka = kk * 32 + kh * 16;
        v16h a  = *(const v16h*)(arow1 + ka);
        v16h b  = *(const v16h*)(brow + LSTM_I + kk * 32 + kh * 16);
        acc = __builtin_amdgcn_wmma_f32_16x16x32_f16(false, a, false, b,
                                                     (short)0, acc, false, false);
    }

    // D layout: VGPR r, lanes 0-15 -> row r, col lane; lanes 16-31 -> row r+8.
    const float* bptr = (wave == 0) ? bF : (wave == 1) ? bI : (wave == 2) ? bC : bO;
    const float bias  = bptr[n];

    __shared__ float gates[4][256];
#pragma unroll
    for (int r = 0; r < 8; ++r)
        gates[wave][(r + 8 * kh) * 16 + l15] = acc[r] + bias;

    __syncthreads();

    // Fused cell update: 128 threads x 2 elements of the 16x16 tile.
    for (int e = threadIdx.x; e < 256; e += 128) {
        const int md  = e >> 4;
        const int col = e & 15;
        const float fg = gates[0][e];
        const float ig = gates[1][e];
        const float cg = gates[2][e];
        const float og = gates[3][e];
        const float fs = 1.0f / (1.0f + __expf(-fg));
        const float is = 1.0f / (1.0f + __expf(-ig));
        const float os = 1.0f / (1.0f + __expf(-og));
        const float ch = tanhf(cg);

        const int b    = mt * 16 + md;
        const int hcol = nt * 16 + col;
        const size_t idx = (size_t)b * LSTM_H + hcol;

        const float cn = fs * cState[idx] + is * ch;
        const float hn = os * tanhf(cn);
        cState[idx]  = cn;
        hOutF32[idx] = hn;
        hOutF16[idx] = (_Float16)hn;
        if (seqOut)
            seqOut[(size_t)b * (LSTM_T * (size_t)LSTM_H) + (size_t)t * LSTM_H + hcol] = hn;
    }
}

// ---------------- host driver ----------------

extern "C" void kernel_launch(void* const* d_in, const int* in_sizes, int n_in,
                              void* d_out, int out_size, void* d_ws, size_t ws_size,
                              hipStream_t stream) {
    (void)in_sizes; (void)n_in; (void)out_size; (void)ws_size;

    const float* x  = (const float*)d_in[0];
    const float* Wg[4] = { (const float*)d_in[1], (const float*)d_in[3],
                           (const float*)d_in[5], (const float*)d_in[7] };  // f,i,c,o
    const float* bg[4] = { (const float*)d_in[2], (const float*)d_in[4],
                           (const float*)d_in[6], (const float*)d_in[8] };
    float* out = (float*)d_out;
    char*  ws  = (char*)d_ws;

    const size_t WPACK_ELEMS = 4ull * LSTM_L * LSTM_H * LSTM_K;   // 16.78M f16
    const size_t XH_ELEMS    = (size_t)LSTM_B * LSTM_T * LSTM_I;  // 16.78M f16
    const size_t BH          = (size_t)LSTM_B * LSTM_H;           // 65536

    _Float16* wp = (_Float16*)ws;                                  // [g][l][H][K]
    _Float16* xh = (_Float16*)(ws + WPACK_ELEMS * 2);              // [B][T][I]
    _Float16* hb = (_Float16*)(ws + (WPACK_ELEMS + XH_ELEMS) * 2); // [l][parity][B*H]
    float*    cs = (float*)((char*)hb + LSTM_L * 2 * BH * 2);      // [l][B*H]
    float*    hs = (float*)((char*)cs + LSTM_L * BH * 4);          // [l][B*H]

    // --- prep: f16 weights/x, zero h(f16 both parities) + c + h(f32) ---
    for (int g = 0; g < 4; ++g)
        f32_to_f16_kernel<<<2048, 256, 0, stream>>>(
            Wg[g], wp + (size_t)g * LSTM_L * LSTM_H * LSTM_K,
            LSTM_L * LSTM_H * LSTM_K);
    f32_to_f16_kernel<<<4096, 256, 0, stream>>>(x, xh, (int)XH_ELEMS);
    {   // hb (512KB) + cs (512KB) + hs (512KB) contiguous -> zero as u32
        const int words = (int)((LSTM_L * 2 * BH * 2 + LSTM_L * BH * 4 * 2) / 4);
        zero_u32_kernel<<<512, 256, 0, stream>>>((unsigned*)hb, words);
    }

    const size_t gateStrideL = (size_t)LSTM_H * LSTM_K;  // per-layer offset in wp
    dim3 grid(LSTM_H / 16, LSTM_B / 16);                 // (64, 4)

    for (int t = 0; t < LSTM_T; ++t) {
        const int p = t & 1;
        _Float16* h0r = hb + 0 * (2 * BH) + (size_t)p       * BH;
        _Float16* h0w = hb + 0 * (2 * BH) + (size_t)(1 - p) * BH;
        _Float16* h1r = hb + 1 * (2 * BH) + (size_t)p       * BH;
        _Float16* h1w = hb + 1 * (2 * BH) + (size_t)(1 - p) * BH;

        // layer 0: A = [ x_t | h0_{t-1} ]
        lstm_step_kernel<<<grid, 128, 0, stream>>>(
            xh + (size_t)t * LSTM_I, (size_t)LSTM_T * LSTM_I, h0r,
            wp /* layer 0 */,
            bg[0], bg[1], bg[2], bg[3],
            cs, hs, h0w, nullptr, t);

        // layer 1: A = [ h0_t | h1_{t-1} ]
        lstm_step_kernel<<<grid, 128, 0, stream>>>(
            h0w, (size_t)LSTM_H, h1r,
            wp + gateStrideL /* layer 1 */,
            bg[0] + LSTM_H, bg[1] + LSTM_H, bg[2] + LSTM_H, bg[3] + LSTM_H,
            cs + BH, hs + BH, h1w, out, t);
    }

    // tail of d_out: ht [L,B,H] then ct [L,B,H]
    copy_states_kernel<<<512, 256, 0, stream>>>(
        hs, cs, out + (size_t)LSTM_B * LSTM_T * LSTM_H, (int)(LSTM_L * BH));
}
